// GPT_76785425318630
// MI455X (gfx1250) — matrix operand order repaired and checked by reading, not compiled, not run on hardware
//
#include <hip/hip_runtime.h>
#include <hip/hip_bf16.h>

// ---------------- model dims ----------------
#define DMODEL 768
#define NHEAD  12
#define HDIM   64
#define NLAYER 12
#define FFN    3072
#define BATCH  4
#define SEQ    1024
#define ROWS   (BATCH*SEQ)   // 4096
#define VOCAB  50257

typedef __attribute__((ext_vector_type(16))) _Float16     v16h;
typedef __attribute__((ext_vector_type(8)))  float        v8f;
typedef __attribute__((ext_vector_type(4)))  unsigned int v4u;

union H16 { v16h v; _Float16 h[16]; unsigned short u[16]; v4u q[2]; };

__device__ __forceinline__ unsigned short f2h(float f) {
  _Float16 h = (_Float16)f;
  return __builtin_bit_cast(unsigned short, h);
}

// ---------------- Tensor Data Mover (CDNA5 TDM) ----------------
#if defined(__AMDGCN__) && __has_builtin(__builtin_amdgcn_tensor_load_to_lds) && \
    __has_builtin(__builtin_amdgcn_s_wait_tensorcnt)
#define USE_TDM 1
#else
#define USE_TDM 0
#endif

#if USE_TDM
typedef __attribute__((ext_vector_type(4))) unsigned int u32x4;
typedef __attribute__((ext_vector_type(8))) int          i32x8;
typedef __attribute__((ext_vector_type(4))) int          i32x4;

// DMA a 2D f16 tile [tile_rows x tile_cols] (row stride row_stride elems,
// tensor bounds tensor_cols x tensor_rows for OOB zero-fill) into LDS.
// D# layout per CDNA5 ISA: group0 = count/lds/global/type,
// group1 = data_size + dims + tile dims + dim0 stride. Remaining groups zero (2D).
// This toolchain exposes the 6-arg builtin form:
//   (u32x4 g0, i32x8 g1, i32x4 g2, i32x4 g3, i32x8 extra, i32 cpol)
__device__ __forceinline__ void tdm_load_tile_f16(
    const unsigned short* gsrc, unsigned short* lds_dst,
    unsigned tile_cols, unsigned tile_rows,
    unsigned tensor_cols, unsigned tensor_rows, unsigned row_stride) {
  unsigned long long ga = (unsigned long long)gsrc;
  unsigned lds_off = (unsigned)(unsigned long long)lds_dst;   // LDS byte offset
  u32x4 g0;
  g0[0] = 1u;                                                  // count=1, user D#
  g0[1] = lds_off;                                             // lds_addr (bytes)
  g0[2] = (unsigned)(ga & 0xFFFFFFFFull);                      // global_addr[31:0]
  g0[3] = (unsigned)((ga >> 32) & 0x1FFFFFFull) | (2u << 30);  // addr[56:32] | type=2
  i32x8 g1;
  g1[0] = (int)(1u << 16);                                     // data_size=1 -> 2 bytes
  g1[1] = (int)((tensor_cols & 0xFFFFu) << 16);                // tensor_dim0[15:0]
  g1[2] = (int)((tensor_cols >> 16) | ((tensor_rows & 0xFFFFu) << 16)); // d0 hi | d1 lo
  g1[3] = (int)((tensor_rows >> 16) | (tile_cols << 16));      // d1 hi | tile_dim0
  g1[4] = (int)tile_rows;                                      // tile_dim1 (tile_dim2=0)
  g1[5] = (int)row_stride;                                     // tensor_dim0_stride lo
  g1[6] = 0;                                                   // stride hi | dim1_stride lo
  g1[7] = 0;
  i32x4 z4 = {0, 0, 0, 0};
  i32x8 z8 = {0, 0, 0, 0, 0, 0, 0, 0};
  __builtin_amdgcn_tensor_load_to_lds(g0, g1, z4, z4, z8, 0);
}
#endif

// ---------------- fp32 -> fp16 convert ----------------
__global__ void k_f32_to_f16(const float* __restrict__ a,
                             unsigned short* __restrict__ o, int n) {
  int i = blockIdx.x * 256 + threadIdx.x;
  if (i < n) o[i] = f2h(a[i]);
}

// ---------------- embedding: x = wte[idx] + wpe[t] ----------------
__global__ void k_embed(const int* __restrict__ idx, const float* __restrict__ wte,
                        const float* __restrict__ wpe, float* __restrict__ x) {
  int i  = blockIdx.x * 256 + threadIdx.x;   // exactly ROWS*DMODEL threads
  int bt = i / DMODEL;
  int d  = i - bt * DMODEL;
  int t  = bt & (SEQ - 1);
  x[i] = wte[(size_t)idx[bt] * DMODEL + d] + wpe[(size_t)t * DMODEL + d];
}

// ---------------- layernorm (fp32 in, f16 out) ----------------
__global__ __launch_bounds__(256)
void k_layernorm(const float* __restrict__ x, const float* __restrict__ w,
                 const float* __restrict__ b, unsigned short* __restrict__ o) {
  __shared__ float s1[256], s2[256];
  const int row = blockIdx.x, tid = threadIdx.x;
  const float* xr = x + (size_t)row * DMODEL;
  float v0 = xr[tid], v1 = xr[tid + 256], v2 = xr[tid + 512];
  s1[tid] = v0 + v1 + v2;
  s2[tid] = v0 * v0 + v1 * v1 + v2 * v2;
  __syncthreads();
  for (int st = 128; st > 0; st >>= 1) {
    if (tid < st) { s1[tid] += s1[tid + st]; s2[tid] += s2[tid + st]; }
    __syncthreads();
  }
  float mean = s1[0] * (1.0f / DMODEL);
  float var  = s2[0] * (1.0f / DMODEL) - mean * mean;
  float rs   = rsqrtf(var + 1e-5f);
  unsigned short* orow = o + (size_t)row * DMODEL;
  orow[tid]       = f2h((v0 - mean) * rs * w[tid]       + b[tid]);
  orow[tid + 256] = f2h((v1 - mean) * rs * w[tid + 256] + b[tid + 256]);
  orow[tid + 512] = f2h((v2 - mean) * rs * w[tid + 512] + b[tid + 512]);
}

// ---------------- WMMA GEMM ----------------
// C[M,N] = A[M,K](f16,rm) x B + bias (+gelu) (+residual)
// TRANSB=0: B is [K,N] row-major (N must be multiple of 128)
// TRANSB=1: B is [N,K] row-major (lm_head; N bounds-checked / TDM zero-fill)
template<bool TRANSB, bool BIAS, bool GELU, bool RESID, bool OUT16>
__global__ __launch_bounds__(256)
void k_gemm(const unsigned short* __restrict__ A,
            const unsigned short* __restrict__ Bm,
            const float* __restrict__ bias,
            const float* __restrict__ Res,
            float* __restrict__ Cf, unsigned short* __restrict__ Ch,
            int M, int N, int K) {
  __shared__ unsigned short As[128 * 32];   // [m][k]
  __shared__ unsigned short Bs[128 * 32];   // [n][k]
  const int tid  = threadIdx.x;
  const int lane = tid & 31, wave = tid >> 5;
  const int wm = wave >> 2, wn = wave & 3;         // 2x4 wave grid
  const int half = lane >> 4, lr = lane & 15;
  const int blockN = blockIdx.x * 128, blockM = blockIdx.y * 128;

  v8f acc[4][2] = {};

  for (int kb = 0; kb < K; kb += 32) {
#if USE_TDM
    if (tid == 0) {   // one TDM descriptor per tile; tracked by TENSORcnt
      tdm_load_tile_f16(A + (size_t)blockM * K + kb, As,
                        32u, 128u, (unsigned)(K - kb), (unsigned)(M - blockM),
                        (unsigned)K);
      if (TRANSB)
        tdm_load_tile_f16(Bm + (size_t)blockN * K + kb, Bs,
                          32u, 128u, (unsigned)(K - kb), (unsigned)(N - blockN),
                          (unsigned)K);
    }
#else
    { // A tile: 2 threads/row, 16 contiguous halves each
      int row = tid >> 1, ch = tid & 1;
      const unsigned short* srcp = A + (size_t)(blockM + row) * K + kb + ch * 16;
      const v4u* src = (const v4u*)srcp;
      v4u* dst = (v4u*)(As + row * 32 + ch * 16);
      dst[0] = src[0]; dst[1] = src[1];
      if (kb + 32 < K) __builtin_prefetch(srcp + 32, 0, 1);
    }
    if (TRANSB) { // rows of W[N,K] copied straight in
      int n = tid >> 1, ch = tid & 1;
      v4u* dst = (v4u*)(Bs + n * 32 + ch * 16);
      if (blockN + n < N) {
        const unsigned short* srcp = Bm + (size_t)(blockN + n) * K + kb + ch * 16;
        const v4u* src = (const v4u*)srcp;
        dst[0] = src[0]; dst[1] = src[1];
        if (kb + 32 < K) __builtin_prefetch(srcp + 32, 0, 1);
      } else {
        v4u z = {0u, 0u, 0u, 0u};
        dst[0] = z; dst[1] = z;
      }
    }
#endif
    if (!TRANSB) {  // transpose [K,N] rows into Bs[n][k]
      int k = tid >> 3, nc = (tid & 7) * 16;
      const unsigned short* src = Bm + (size_t)(kb + k) * N + blockN + nc;
      unsigned short tmp[16];
      *(v4u*)&tmp[0] = ((const v4u*)src)[0];
      *(v4u*)&tmp[8] = ((const v4u*)src)[1];
      #pragma unroll
      for (int j = 0; j < 16; j++) Bs[(nc + j) * 32 + k] = tmp[j];
      if (kb + 32 < K) __builtin_prefetch(src + (size_t)32 * N, 0, 1);
    }
#if USE_TDM
    if (tid == 0) __builtin_amdgcn_s_wait_tensorcnt(0);
#endif
    __syncthreads();

    H16 bf[2];
    #pragma unroll
    for (int sn = 0; sn < 2; sn++) {
      const v4u* p = (const v4u*)(Bs + (wn * 32 + sn * 16 + lr) * 32 + half * 16);
      bf[sn].q[0] = p[0]; bf[sn].q[1] = p[1];
    }
    #pragma unroll
    for (int sm = 0; sm < 4; sm++) {
      H16 af;
      const unsigned short* ap = As + (wm * 64 + sm * 16 + lr) * 32;
      af.q[0] = *(const v4u*)(ap + half * 16);
      af.q[1] = *(const v4u*)(ap + half * 16 + 8);
      #pragma unroll
      for (int sn = 0; sn < 2; sn++)
        acc[sm][sn] = __builtin_amdgcn_wmma_f32_16x16x32_f16(
            false, af.v, false, bf[sn].v, (short)0, acc[sm][sn], false, false);
    }
    __syncthreads();
  }

  #pragma unroll
  for (int sm = 0; sm < 4; sm++)
    #pragma unroll
    for (int sn = 0; sn < 2; sn++)
      #pragma unroll
      for (int r = 0; r < 8; r++) {
        int m = blockM + wm * 64 + sm * 16 + r + half * 8;
        int n = blockN + wn * 32 + sn * 16 + lr;
        if (n < N) {
          float v = acc[sm][sn][r];
          if (BIAS)  v += bias[n];
          if (GELU)  v = 0.5f * v * (1.0f + erff(v * 0.70710678f));
          if (RESID) v += Res[(size_t)m * N + n];
          if (OUT16) Ch[(size_t)m * N + n] = f2h(v);
          else       Cf[(size_t)m * N + n] = v;
        }
      }
}

// ---------------- fused causal attention (flash-style) ----------------
// one wave handles 16 query rows of one (b, head); qkv f16 [ROWS, 3*DMODEL]
__global__ __launch_bounds__(32)
void k_attention(const unsigned short* __restrict__ qkv,
                 unsigned short* __restrict__ y) {
  __shared__ unsigned short Pt[16 * 32];   // P tile [16q][32k]
  __shared__ unsigned short Vt[32 * 64];   // V tile [32k][64hd]
  const int lane = threadIdx.x;
  const int half = lane >> 4, lr = lane & 15;
  const int qbase = blockIdx.x * 16;
  const int hh = blockIdx.y, b = blockIdx.z;
  const size_t rstride = 3 * DMODEL;

  H16 qa[2];  // Q A-fragments, hd chunks 0..31 / 32..63
  {
    const unsigned short* qrow =
        qkv + ((size_t)(b * SEQ + qbase + lr)) * rstride + hh * HDIM;
    #pragma unroll
    for (int c = 0; c < 2; c++) {
      qa[c].q[0] = *(const v4u*)(qrow + c * 32 + half * 8);
      qa[c].q[1] = *(const v4u*)(qrow + c * 32 + 16 + half * 8);
    }
  }

  v8f of[4] = {};
  float mrow[8], lrow[8];
  #pragma unroll
  for (int r = 0; r < 8; r++) { mrow[r] = -1e30f; lrow[r] = 0.0f; }

  const float scale = 0.125f;              // 1/sqrt(64)
  const int nblocks = ((qbase + 15) >> 5) + 1;

  for (int kb = 0; kb < nblocks; kb++) {
    const int kbase = kb * 32;
    float sv[2][8];
    #pragma unroll
    for (int sn = 0; sn < 2; sn++) {       // S = Q K^T for 16 keys
      const int kn = kbase + sn * 16;
      const unsigned short* krow =
          qkv + ((size_t)(b * SEQ + kn + lr)) * rstride + DMODEL + hh * HDIM;
      v8f s = {};
      #pragma unroll
      for (int c = 0; c < 2; c++) {
        H16 kf;
        kf.q[0] = *(const v4u*)(krow + c * 32 + half * 8);
        kf.q[1] = *(const v4u*)(krow + c * 32 + 16 + half * 8);
        s = __builtin_amdgcn_wmma_f32_16x16x32_f16(
            false, qa[c].v, false, kf.v, (short)0, s, false, false);
      }
      const int keycol = kn + lr;
      #pragma unroll
      for (int r = 0; r < 8; r++) {
        int q = qbase + r + half * 8;
        float xx = s[r] * scale;
        sv[sn][r] = (keycol <= q) ? xx : -1e30f;   // causal mask
      }
    }
    // online softmax update (row = across 16 lanes of a half-wave)
    #pragma unroll
    for (int r = 0; r < 8; r++) {
      float rm = fmaxf(sv[0][r], sv[1][r]);
      rm = fmaxf(rm, __shfl_xor(rm, 1));
      rm = fmaxf(rm, __shfl_xor(rm, 2));
      rm = fmaxf(rm, __shfl_xor(rm, 4));
      rm = fmaxf(rm, __shfl_xor(rm, 8));
      float mnew = fmaxf(mrow[r], rm);
      float cf = expf(mrow[r] - mnew);
      mrow[r] = mnew;
      float p0 = expf(sv[0][r] - mnew);
      float p1 = expf(sv[1][r] - mnew);
      float ps = p0 + p1;
      ps += __shfl_xor(ps, 1);
      ps += __shfl_xor(ps, 2);
      ps += __shfl_xor(ps, 4);
      ps += __shfl_xor(ps, 8);
      lrow[r] = lrow[r] * cf + ps;
      #pragma unroll
      for (int nt = 0; nt < 4; nt++) of[nt][r] *= cf;
      int prow = r + half * 8;
      Pt[prow * 32 + lr]      = f2h(p0);   // C-layout -> LDS
      Pt[prow * 32 + 16 + lr] = f2h(p1);
    }
    { // stage V tile [32 keys][64 hd]
      const unsigned short* vrow =
          qkv + ((size_t)(b * SEQ + kbase + lane)) * rstride + 2 * DMODEL + hh * HDIM;
      v4u* dst = (v4u*)(Vt + lane * 64);
      const v4u* src = (const v4u*)vrow;
      #pragma unroll
      for (int j = 0; j < 8; j++) dst[j] = src[j];
    }
    asm volatile("s_wait_dscnt 0" ::: "memory");
    // O += P @ V
    H16 pa;
    pa.q[0] = *(const v4u*)(Pt + lr * 32 + half * 8);
    pa.q[1] = *(const v4u*)(Pt + lr * 32 + 16 + half * 8);
    #pragma unroll
    for (int nt = 0; nt < 4; nt++) {
      H16 vf;
      #pragma unroll
      for (int t = 0; t < 16; t++)
        vf.u[t] = Vt[(half * 8 + t + (t >= 8 ? 8 : 0)) * 64 + nt * 16 + lr];
      of[nt] = __builtin_amdgcn_wmma_f32_16x16x32_f16(
          false, pa.v, false, vf.v, (short)0, of[nt], false, false);
    }
    asm volatile("s_wait_dscnt 0" ::: "memory");
  }

  #pragma unroll
  for (int r = 0; r < 8; r++) {
    float inv = 1.0f / lrow[r];
    int q = qbase + r + half * 8;
    unsigned short* yrow = y + ((size_t)(b * SEQ + q)) * DMODEL + hh * HDIM;
    #pragma unroll
    for (int nt = 0; nt < 4; nt++)
      yrow[nt * 16 + lr] = f2h(of[nt][r] * inv);
  }
}

// ---------------- host orchestration ----------------
extern "C" void kernel_launch(void* const* d_in, const int* in_sizes, int n_in,
                              void* d_out, int out_size, void* d_ws, size_t ws_size,
                              hipStream_t stream) {
  const int*   idx  = (const int*)  d_in[0];
  const float* wte  = (const float*)d_in[1];
  const float* wpe  = (const float*)d_in[2];
  const float* ln1w = (const float*)d_in[3];
  const float* ln1b = (const float*)d_in[4];
  const float* qkvw = (const float*)d_in[5];
  const float* qkvb = (const float*)d_in[6];
  const float* apw  = (const float*)d_in[7];
  const float* apb  = (const float*)d_in[8];
  const float* ln2w = (const float*)d_in[9];
  const float* ln2b = (const float*)d_in[10];
  const float* fcw  = (const float*)d_in[11];
  const float* fcb  = (const float*)d_in[12];
  const float* mpw  = (const float*)d_in[13];
  const float* mpb  = (const float*)d_in[14];
  const float* lnfw = (const float*)d_in[15];
  const float* lnfb = (const float*)d_in[16];

  char* ws = (char*)d_ws;
  size_t off = 0;
  auto alloc = [&](size_t bytes) -> char* {
    char* p = ws + off;
    off += (bytes + 255) & ~(size_t)255;
    return p;
  };
  float*          x     = (float*)         alloc((size_t)ROWS * DMODEL * 4);
  unsigned short* hbuf  = (unsigned short*)alloc((size_t)ROWS * DMODEL * 2);
  unsigned short* qkvB  = (unsigned short*)alloc((size_t)ROWS * 3 * DMODEL * 2);
  unsigned short* ybuf  = (unsigned short*)alloc((size_t)ROWS * DMODEL * 2);
  unsigned short* act   = (unsigned short*)alloc((size_t)ROWS * FFN * 2);
  unsigned short* wte16 = (unsigned short*)alloc((size_t)VOCAB * DMODEL * 2);
  unsigned short* qw16  = (unsigned short*)alloc((size_t)NLAYER * DMODEL * 3 * DMODEL * 2);
  unsigned short* aw16  = (unsigned short*)alloc((size_t)NLAYER * DMODEL * DMODEL * 2);
  unsigned short* fw16  = (unsigned short*)alloc((size_t)NLAYER * DMODEL * FFN * 2);
  unsigned short* mw16  = (unsigned short*)alloc((size_t)NLAYER * FFN * DMODEL * 2);

  auto conv = [&](const float* s, unsigned short* d, size_t n) {
    k_f32_to_f16<<<(unsigned)((n + 255) / 256), 256, 0, stream>>>(s, d, (int)n);
  };
  conv(wte,  wte16, (size_t)VOCAB * DMODEL);
  conv(qkvw, qw16,  (size_t)NLAYER * DMODEL * 3 * DMODEL);
  conv(apw,  aw16,  (size_t)NLAYER * DMODEL * DMODEL);
  conv(fcw,  fw16,  (size_t)NLAYER * DMODEL * FFN);
  conv(mpw,  mw16,  (size_t)NLAYER * FFN * DMODEL);

  k_embed<<<(ROWS * DMODEL) / 256, 256, 0, stream>>>(idx, wte, wpe, x);

  const dim3 blk(256);
  for (int l = 0; l < NLAYER; l++) {
    // ln1 -> h
    k_layernorm<<<ROWS, blk, 0, stream>>>(x, ln1w + l * DMODEL, ln1b + l * DMODEL, hbuf);
    // qkv = h @ qkv_w + qkv_b  (f16 out)
    k_gemm<false, true, false, false, true>
        <<<dim3(3 * DMODEL / 128, ROWS / 128), blk, 0, stream>>>(
            hbuf, qw16 + (size_t)l * DMODEL * 3 * DMODEL, qkvb + (size_t)l * 3 * DMODEL,
            nullptr, nullptr, qkvB, ROWS, 3 * DMODEL, DMODEL);
    // attention -> y (f16)
    k_attention<<<dim3(SEQ / 16, NHEAD, BATCH), dim3(32), 0, stream>>>(qkvB, ybuf);
    // x = x + y @ attn_proj_w + attn_proj_b
    k_gemm<false, true, false, true, false>
        <<<dim3(DMODEL / 128, ROWS / 128), blk, 0, stream>>>(
            ybuf, aw16 + (size_t)l * DMODEL * DMODEL, apb + (size_t)l * DMODEL,
            x, x, nullptr, ROWS, DMODEL, DMODEL);
    // ln2 -> h
    k_layernorm<<<ROWS, blk, 0, stream>>>(x, ln2w + l * DMODEL, ln2b + l * DMODEL, hbuf);
    // act = gelu(h @ fc_w + fc_b)  (f16 out)
    k_gemm<false, true, true, false, true>
        <<<dim3(FFN / 128, ROWS / 128), blk, 0, stream>>>(
            hbuf, fw16 + (size_t)l * DMODEL * FFN, fcb + (size_t)l * FFN,
            nullptr, nullptr, act, ROWS, FFN, DMODEL);
    // x = x + act @ mlp_proj_w + mlp_proj_b
    k_gemm<false, true, false, true, false>
        <<<dim3(DMODEL / 128, ROWS / 128), blk, 0, stream>>>(
            act, mw16 + (size_t)l * FFN * DMODEL, mpb + (size_t)l * DMODEL,
            x, x, nullptr, ROWS, DMODEL, FFN);
  }

  // final LN -> h; logits = h @ wte^T  (fp32 out)
  k_layernorm<<<ROWS, blk, 0, stream>>>(x, lnfw, lnfb, hbuf);
  k_gemm<true, false, false, false, false>
      <<<dim3((VOCAB + 127) / 128, ROWS / 128), blk, 0, stream>>>(
          hbuf, wte16, nullptr, nullptr, (float*)d_out, nullptr, ROWS, VOCAB, DMODEL);
}
